// AttnDecoder_70171175682281
// MI455X (gfx1250) — matrix operand order repaired and checked by reading, not compiled, not run on hardware
//
#include <hip/hip_runtime.h>
#include <math.h>

// Problem sizes (match reference)
constexpr int B  = 2048;
constexpr int T  = 200;
constexpr int H  = 256;
constexpr int FR = 80;
constexpr int P1 = 256;
constexpr int P2 = 128;
constexpr int OUT = FR * 3; // 240

typedef __attribute__((ext_vector_type(8)))  float    v8f;
typedef __attribute__((ext_vector_type(8)))  _Float16 v8h;
typedef __attribute__((ext_vector_type(16))) _Float16 v16h;

__device__ __forceinline__ float sigmoidf_(float x) {
  return 1.0f / (1.0f + expf(-x));
}

__device__ __forceinline__ v16h cat8h(v8h lo, v8h hi) {
  return __builtin_shufflevector(lo, hi, 0, 1, 2, 3, 4, 5, 6, 7, 8, 9, 10, 11,
                                 12, 13, 14, 15);
}

// ---------------------------------------------------------------------------
// Kernel 0: repack W1 (H x H f32, used as B = W1^T) into f16 WMMA B-fragment
// order for V_WMMA_F32_16X16X32_F16.
// B-frag layout used here (contiguous k, same order as the A loads): lane l holds column n = l&15; K values
// 16*(l>>4) + j for j=0..15, packed sequentially (2 per VGPR, v0..v7).
// pack half index: (((ks*16 + ntile)*32 + lane)*16 + j)
//   n = ntile*16 + (lane&15), K = ks*32 + 16*(lane>>4) + j
// ---------------------------------------------------------------------------
__global__ __launch_bounds__(256) void k_pack_w1(const float* __restrict__ w1,
                                                 _Float16* __restrict__ pack) {
  int i = blockIdx.x * blockDim.x + threadIdx.x; // 0 .. 8*16*32*16-1 = 65535
  if (i >= 8 * 16 * 32 * 16) return;
  int j     = i & 15;
  int lane  = (i >> 4) & 31;
  int ntile = (i >> 9) & 15;
  int ks    = i >> 13;
  int n = ntile * 16 + (lane & 15);
  int K = ks * 32 + 16 * (lane >> 4) + j;
  pack[i] = (_Float16)w1[n * H + K];
}

// ---------------------------------------------------------------------------
// Kernel 1: PreNet (fc1+relu, fc2+relu) -> attention GRUCell -> relu ->
//           dec_proj = attn_out @ w2^T.  One block per batch row, 256 thr.
// ---------------------------------------------------------------------------
__global__ __launch_bounds__(256) void k_prenet_attngru(
    const float* __restrict__ inp, const float* __restrict__ attn_h,
    const float* __restrict__ fc1_w, const float* __restrict__ fc1_b,
    const float* __restrict__ fc2_w, const float* __restrict__ fc2_b,
    const float* __restrict__ wih, const float* __restrict__ whh,
    const float* __restrict__ bih, const float* __restrict__ bhh,
    const float* __restrict__ w2,
    float* __restrict__ new_attn_h_out,   // d_out region
    float* __restrict__ attn_out_ws,      // ws
    float* __restrict__ dec_proj_ws) {    // ws
  int b = blockIdx.x, j = threadIdx.x;
  __shared__ float xL[FR], hL[H], p1L[P1], p2L[P2], aoL[H];
  if (j < FR) xL[j] = inp[b * FR + j];
  hL[j] = attn_h[b * H + j];
  __syncthreads();

  float acc = fc1_b[j];
  for (int k = 0; k < FR; ++k) acc += xL[k] * fc1_w[j * FR + k];
  p1L[j] = fmaxf(acc, 0.0f);
  __syncthreads();

  if (j < P2) {
    float a2 = fc2_b[j];
    for (int k = 0; k < P1; ++k) a2 += p1L[k] * fc2_w[j * P1 + k];
    p2L[j] = fmaxf(a2, 0.0f);
  }
  __syncthreads();

  float gir = bih[j], giz = bih[H + j], gin = bih[2 * H + j];
  for (int k = 0; k < P2; ++k) {
    float x = p2L[k];
    gir += x * wih[j * P2 + k];
    giz += x * wih[(H + j) * P2 + k];
    gin += x * wih[(2 * H + j) * P2 + k];
  }
  float ghr = bhh[j], ghz = bhh[H + j], ghn = bhh[2 * H + j];
  for (int k = 0; k < H; ++k) {
    float x = hL[k];
    ghr += x * whh[j * H + k];
    ghz += x * whh[(H + j) * H + k];
    ghn += x * whh[(2 * H + j) * H + k];
  }
  float r = sigmoidf_(gir + ghr);
  float z = sigmoidf_(giz + ghz);
  float nst = tanhf(gin + r * ghn);
  float hn = (1.0f - z) * nst + z * hL[j];
  new_attn_h_out[(size_t)b * H + j] = hn;
  float ao = fmaxf(hn, 0.0f);
  attn_out_ws[(size_t)b * H + j] = ao;
  aoL[j] = ao;
  __syncthreads();

  float dp = 0.0f;
  for (int k = 0; k < H; ++k) dp += aoL[k] * w2[j * H + k];
  dec_proj_ws[(size_t)b * H + j] = dp;
}

// ---------------------------------------------------------------------------
// Kernel 2 (the big one): u[b,t] = v . tanh(W1 @ enc[b,t] + dec_proj[b])
// f16 WMMA 16x16x32 with f32 accumulation.
// Block = 256 thr (8 waves), 32 t-rows of one b, all 256 columns.
// Wave: 2 row-subtiles x 2 col-subtiles (4 v8f accumulators), K-loop of 8.
// enc tile converted f32->f16 during LDS staging.
// ---------------------------------------------------------------------------
constexpr int RT    = 32;  // rows per block
constexpr int ELDPH = 264; // LDS row stride in halves (16B aligned, spreads banks)

__global__ __launch_bounds__(256) void k_attn_scores(
    const float* __restrict__ enc, const _Float16* __restrict__ w1pack,
    const float* __restrict__ dec_proj, const float* __restrict__ vvec,
    float* __restrict__ u) {
  const int b   = blockIdx.y;
  const int t0  = blockIdx.x * RT;
  const int tid = threadIdx.x;

  __shared__ _Float16 encL[RT * ELDPH];
  __shared__ float dpL[H];
  __shared__ float vL[H];
  __shared__ float upart[8][RT];

  // Stage enc tile: RT rows x 256 cols, coalesced f32 loads, f16 LDS stores.
  for (int rr = 0; rr < RT; ++rr) {
    int t = t0 + rr;
    float val = (t < T) ? enc[((size_t)b * T + t) * H + tid] : 0.0f;
    encL[rr * ELDPH + tid] = (_Float16)val;
  }
  dpL[tid] = dec_proj[(size_t)b * H + tid];
  vL[tid]  = vvec[tid];
  __syncthreads();

  const int lane = tid & 31;
  const int wave = tid >> 5;
  const int half = lane >> 4;  // 0: lanes 0-15, 1: lanes 16-31
  const int mrow = lane & 15;

  v8f c00 = {}, c01 = {}, c10 = {}, c11 = {};

  // A rows for the two row-subtiles handled by this lane.
  const v8h* arow0 = (const v8h*)(encL + mrow * ELDPH);
  const v8h* arow1 = (const v8h*)(encL + (16 + mrow) * ELDPH);
  const v8h* bp    = (const v8h*)w1pack;

#pragma unroll
  for (int ks = 0; ks < 8; ++ks) {
    // A frag (contiguous, matches B pack): halves [ks*32 + 16*half .. +15].
    v8h a0lo = arow0[ks * 4 + 2 * half];
    v8h a0hi = arow0[ks * 4 + 2 * half + 1];
    v8h a1lo = arow1[ks * 4 + 2 * half];
    v8h a1hi = arow1[ks * 4 + 2 * half + 1];
    v16h a0 = cat8h(a0lo, a0hi);
    v16h a1 = cat8h(a1lo, a1hi);
    // B frags: 2x coalesced 16B from packed W1 (ntile = 2*wave, 2*wave+1).
    int bb = ((ks * 16 + 2 * wave) * 32 + lane) * 2;
    v16h b0 = cat8h(bp[bb], bp[bb + 1]);
    v16h b1 = cat8h(bp[bb + 64], bp[bb + 65]);

    c00 = __builtin_amdgcn_wmma_f32_16x16x32_f16(false, a0, false, b0,
                                                 (short)0, c00, false, false);
    c01 = __builtin_amdgcn_wmma_f32_16x16x32_f16(false, a0, false, b1,
                                                 (short)0, c01, false, false);
    c10 = __builtin_amdgcn_wmma_f32_16x16x32_f16(false, a1, false, b0,
                                                 (short)0, c10, false, false);
    c11 = __builtin_amdgcn_wmma_f32_16x16x32_f16(false, a1, false, b1,
                                                 (short)0, c11, false, false);
  }

  // Epilogue: u_partial = v[n]*tanh(S + dec_proj[n]), reduce over n.
  // Lane columns: n1 = 32*wave + mrow (ct=0), n2 = n1 + 16 (ct=1).
  // Lane rows: m = rt*16 + r + 8*half.
  int n1 = wave * 32 + mrow;
  int n2 = n1 + 16;
  float dp1 = dpL[n1], dp2 = dpL[n2];
  float v1 = vL[n1], v2 = vL[n2];
  float part[16];
#pragma unroll
  for (int r = 0; r < 8; ++r) {
    part[r]     = v1 * tanhf(c00[r] + dp1) + v2 * tanhf(c01[r] + dp2);
    part[8 + r] = v1 * tanhf(c10[r] + dp1) + v2 * tanhf(c11[r] + dp2);
  }
  // Butterfly over the 16 lanes sharing the same row set (lane bits 0..3).
#pragma unroll
  for (int m = 1; m < 16; m <<= 1) {
#pragma unroll
    for (int r = 0; r < 16; ++r) part[r] += __shfl_xor(part[r], m, 32);
  }
  if (mrow == 0) {
#pragma unroll
    for (int r = 0; r < 8; ++r) {
      upart[wave][r + 8 * half]      = part[r];
      upart[wave][16 + r + 8 * half] = part[8 + r];
    }
  }
  __syncthreads();
  if (tid < RT) {
    float s = 0.0f;
#pragma unroll
    for (int w = 0; w < 8; ++w) s += upart[w][tid];
    int t = t0 + tid;
    if (t < T) u[(size_t)b * T + t] = s;
  }
}

// ---------------------------------------------------------------------------
// Kernel 3: softmax over T per batch row -> a (written straight to d_out)
// ---------------------------------------------------------------------------
__global__ __launch_bounds__(256) void k_softmax_T(const float* __restrict__ u,
                                                   float* __restrict__ a_out) {
  int b = blockIdx.x, tid = threadIdx.x;
  __shared__ float red[256];
  float val = (tid < T) ? u[(size_t)b * T + tid] : -3.0e38f;
  red[tid] = val;
  __syncthreads();
  for (int s = 128; s > 0; s >>= 1) {
    if (tid < s) red[tid] = fmaxf(red[tid], red[tid + s]);
    __syncthreads();
  }
  float mx = red[0];
  __syncthreads();
  float e = (tid < T) ? expf(val - mx) : 0.0f;
  red[tid] = e;
  __syncthreads();
  for (int s = 128; s > 0; s >>= 1) {
    if (tid < s) red[tid] += red[tid + s];
    __syncthreads();
  }
  float inv = 1.0f / red[0];
  if (tid < T) a_out[(size_t)b * T + tid] = e * inv;
}

// ---------------------------------------------------------------------------
// Kernel 4: ctx[b,h] = sum_t a[b,t] * enc[b,t,h]  (coalesced over h per t)
// ---------------------------------------------------------------------------
__global__ __launch_bounds__(256) void k_context(const float* __restrict__ a,
                                                 const float* __restrict__ enc,
                                                 float* __restrict__ ctx) {
  int b = blockIdx.x, j = threadIdx.x;
  __shared__ float aL[T];
  if (j < T) aL[j] = a[(size_t)b * T + j];
  __syncthreads();
  float acc = 0.0f;
  const float* ep = enc + (size_t)b * T * H + j;
  for (int t = 0; t < T; ++t) acc += aL[t] * ep[(size_t)t * H];
  ctx[(size_t)b * H + j] = acc;
}

// ---------------------------------------------------------------------------
// Kernel 5: combine -> decoder GRU0 (+residual relu) -> GRU1 (+residual relu)
//           -> out linear -> softmax over OUT. One block per batch row.
// ---------------------------------------------------------------------------
__global__ __launch_bounds__(256) void k_decoder(
    const float* __restrict__ attn_out_ws, const float* __restrict__ ctx_ws,
    const float* __restrict__ dec_h0, const float* __restrict__ dec_h1,
    const float* __restrict__ comb_w, const float* __restrict__ comb_b,
    const float* __restrict__ d0wih, const float* __restrict__ d0whh,
    const float* __restrict__ d0bih, const float* __restrict__ d0bhh,
    const float* __restrict__ d1wih, const float* __restrict__ d1whh,
    const float* __restrict__ d1bih, const float* __restrict__ d1bhh,
    const float* __restrict__ out_w, const float* __restrict__ out_b,
    float* __restrict__ out_sm, float* __restrict__ h0n_out,
    float* __restrict__ h1n_out) {
  int b = blockIdx.x, j = threadIdx.x;
  __shared__ float aoL[H], ctxL[H], dinL[H], h0L[H], h1L[H], doL[H], do2L[H];
  __shared__ float red[256];
  aoL[j]  = attn_out_ws[(size_t)b * H + j];
  ctxL[j] = ctx_ws[(size_t)b * H + j];
  h0L[j]  = dec_h0[(size_t)b * H + j];
  h1L[j]  = dec_h1[(size_t)b * H + j];
  __syncthreads();

  float acc = comb_b[j];
  for (int k = 0; k < H; ++k) acc += aoL[k] * comb_w[j * (2 * H) + k];
  for (int k = 0; k < H; ++k) acc += ctxL[k] * comb_w[j * (2 * H) + H + k];
  dinL[j] = acc;
  __syncthreads();

  // GRU0
  {
    float gir = d0bih[j], giz = d0bih[H + j], gin = d0bih[2 * H + j];
    for (int k = 0; k < H; ++k) {
      float x = dinL[k];
      gir += x * d0wih[j * H + k];
      giz += x * d0wih[(H + j) * H + k];
      gin += x * d0wih[(2 * H + j) * H + k];
    }
    float ghr = d0bhh[j], ghz = d0bhh[H + j], ghn = d0bhh[2 * H + j];
    for (int k = 0; k < H; ++k) {
      float x = h0L[k];
      ghr += x * d0whh[j * H + k];
      ghz += x * d0whh[(H + j) * H + k];
      ghn += x * d0whh[(2 * H + j) * H + k];
    }
    float r = sigmoidf_(gir + ghr);
    float z = sigmoidf_(giz + ghz);
    float nst = tanhf(gin + r * ghn);
    float h0n = (1.0f - z) * nst + z * h0L[j];
    h0n_out[(size_t)b * H + j] = h0n;
    doL[j] = dinL[j] + fmaxf(h0n, 0.0f);
  }
  __syncthreads();

  // GRU1
  {
    float gir = d1bih[j], giz = d1bih[H + j], gin = d1bih[2 * H + j];
    for (int k = 0; k < H; ++k) {
      float x = doL[k];
      gir += x * d1wih[j * H + k];
      giz += x * d1wih[(H + j) * H + k];
      gin += x * d1wih[(2 * H + j) * H + k];
    }
    float ghr = d1bhh[j], ghz = d1bhh[H + j], ghn = d1bhh[2 * H + j];
    for (int k = 0; k < H; ++k) {
      float x = h1L[k];
      ghr += x * d1whh[j * H + k];
      ghz += x * d1whh[(H + j) * H + k];
      ghn += x * d1whh[(2 * H + j) * H + k];
    }
    float r = sigmoidf_(gir + ghr);
    float z = sigmoidf_(giz + ghz);
    float nst = tanhf(gin + r * ghn);
    float h1n = (1.0f - z) * nst + z * h1L[j];
    h1n_out[(size_t)b * H + j] = h1n;
    do2L[j] = doL[j] + fmaxf(h1n, 0.0f);
  }
  __syncthreads();

  float lg = -3.0e38f;
  if (j < OUT) {
    lg = out_b[j];
    for (int k = 0; k < H; ++k) lg += do2L[k] * out_w[j * H + k];
  }
  red[j] = lg;
  __syncthreads();
  for (int s = 128; s > 0; s >>= 1) {
    if (j < s) red[j] = fmaxf(red[j], red[j + s]);
    __syncthreads();
  }
  float mx = red[0];
  __syncthreads();
  float e = (j < OUT) ? expf(lg - mx) : 0.0f;
  red[j] = e;
  __syncthreads();
  for (int s = 128; s > 0; s >>= 1) {
    if (j < s) red[j] += red[j + s];
    __syncthreads();
  }
  if (j < OUT) out_sm[(size_t)b * OUT + j] = e / red[0];
}

// ---------------------------------------------------------------------------
extern "C" void kernel_launch(void* const* d_in, const int* in_sizes, int n_in,
                              void* d_out, int out_size, void* d_ws,
                              size_t ws_size, hipStream_t stream) {
  const float* inp      = (const float*)d_in[0];
  const float* attn_h   = (const float*)d_in[1];
  const float* dec_h0   = (const float*)d_in[2];
  const float* dec_h1   = (const float*)d_in[3];
  const float* enc      = (const float*)d_in[4];
  const float* fc1_w    = (const float*)d_in[5];
  const float* fc1_b    = (const float*)d_in[6];
  const float* fc2_w    = (const float*)d_in[7];
  const float* fc2_b    = (const float*)d_in[8];
  const float* attn_wih = (const float*)d_in[9];
  const float* attn_whh = (const float*)d_in[10];
  const float* attn_bih = (const float*)d_in[11];
  const float* attn_bhh = (const float*)d_in[12];
  const float* w1       = (const float*)d_in[13];
  const float* w2       = (const float*)d_in[14];
  const float* vvec     = (const float*)d_in[15];
  const float* comb_w   = (const float*)d_in[16];
  const float* comb_b   = (const float*)d_in[17];
  const float* d0wih    = (const float*)d_in[18];
  const float* d0whh    = (const float*)d_in[19];
  const float* d0bih    = (const float*)d_in[20];
  const float* d0bhh    = (const float*)d_in[21];
  const float* d1wih    = (const float*)d_in[22];
  const float* d1whh    = (const float*)d_in[23];
  const float* d1bih    = (const float*)d_in[24];
  const float* d1bhh    = (const float*)d_in[25];
  const float* out_w    = (const float*)d_in[26];
  const float* out_b    = (const float*)d_in[27];

  float* out   = (float*)d_out;
  float* o_out = out;                         // (B, OUT)
  float* o_nah = o_out + (size_t)B * OUT;     // (B, H)
  float* o_h0n = o_nah + (size_t)B * H;       // (B, H)
  float* o_h1n = o_h0n + (size_t)B * H;       // (B, H)
  float* o_a   = o_h1n + (size_t)B * H;       // (B, T)

  float* ws          = (float*)d_ws;
  _Float16* w1packh  = (_Float16*)ws;               // 65536 halves = 32768 floats
  float* ao_ws   = ws + 32768;                      // B*H
  float* dp_ws   = ao_ws + (size_t)B * H;           // B*H
  float* u_ws    = dp_ws + (size_t)B * H;           // B*T
  float* ctx_ws  = u_ws + (size_t)B * T;            // B*H

  k_pack_w1<<<(8 * 16 * 32 * 16 + 255) / 256, 256, 0, stream>>>(w1, w1packh);

  k_prenet_attngru<<<B, 256, 0, stream>>>(
      inp, attn_h, fc1_w, fc1_b, fc2_w, fc2_b, attn_wih, attn_whh, attn_bih,
      attn_bhh, w2, o_nah, ao_ws, dp_ws);

  dim3 gA((T + RT - 1) / RT, B); // (7, 2048)
  k_attn_scores<<<gA, 256, 0, stream>>>(enc, w1packh, dp_ws, vvec, u_ws);

  k_softmax_T<<<B, 256, 0, stream>>>(u_ws, o_a);

  k_context<<<B, 256, 0, stream>>>(o_a, enc, ctx_ws);

  k_decoder<<<B, 256, 0, stream>>>(ao_ws, ctx_ws, dec_h0, dec_h1, comb_w,
                                   comb_b, d0wih, d0whh, d0bih, d0bhh, d1wih,
                                   d1whh, d1bih, d1bhh, out_w, out_b, o_out,
                                   o_h0n, o_h1n);
}
